// MultiSingleStreamBlockLoraProcessor_38955353375466
// MI455X (gfx1250) — matrix operand order repaired and checked by reading, not compiled, not run on hardware
//
#include <hip/hip_runtime.h>
#include <hip/hip_bf16.h>

// ---------------------------------------------------------------------------
// Problem constants (from reference)
// ---------------------------------------------------------------------------
#define S_TOK   3072
#define DIMC    2560
#define HDC     128
#define NHEADS  20
#define BLOCKQ  2048          // image-token rows (the only rows in the output)
#define EPSF    1e-5f
#define SCALEF  0.08838834764831845f   // 1/sqrt(128)

typedef __bf16 v16bf __attribute__((ext_vector_type(16)));
typedef __bf16 v8bf  __attribute__((ext_vector_type(8)));
typedef float  v8f   __attribute__((ext_vector_type(8)));

__device__ __forceinline__ v8f vzero8() {
    v8f z;
#pragma unroll
    for (int i = 0; i < 8; ++i) z[i] = 0.0f;
    return z;
}

// WMMA wrapper: D = A(16x32 bf16) * B(32x16 bf16) + C(f32)
__device__ __forceinline__ v8f wmma_bf16(v16bf a, v16bf b, v8f c) {
    return __builtin_amdgcn_wmma_f32_16x16x32_bf16(
        /*neg_a=*/false, a, /*neg_b=*/false, b,
        /*c_mod=*/(short)0, c, /*reuse_a=*/false, /*reuse_b=*/false);
}

// A-fragment (16x32, 16-bit), k order chosen here (A and B match): lane l(0..15) holds row M=l with K {koff..koff+15}
// in VGPR0-7, koff = 0 (lanes<16) / 16 (>=16).
// => two contiguous 16B loads per lane.
__device__ __forceinline__ v16bf make_a_frag(const __bf16* p) {
    v8bf lo = *(const v8bf*)p;
    v8bf hh = *(const v8bf*)(p + 8);
    v16bf a;
#pragma unroll
    for (int i = 0; i < 8; ++i) { a[i] = lo[i]; a[i + 8] = hh[i]; }
    return a;
}

__device__ __forceinline__ v16bf make_frag16(const __bf16* p) {
    // 16 contiguous bf16 (for B fragments: lanes<16 hold K 0..15, lanes>=16
    // hold K 16..31; element pair (2v,2v+1) lands in VGPR v). 16B-aligned ok.
    v8bf lo = *(const v8bf*)p;
    v8bf hh = *(const v8bf*)(p + 8);
    v16bf b;
#pragma unroll
    for (int i = 0; i < 8; ++i) { b[i] = lo[i]; b[i + 8] = hh[i]; }
    return b;
}

// ---------------------------------------------------------------------------
// f32 -> bf16 elementwise convert
// ---------------------------------------------------------------------------
__global__ __launch_bounds__(256)
void f32_to_bf16_kernel(const float* __restrict__ src, __bf16* __restrict__ dst, int n) {
    int i = blockIdx.x * 256 + threadIdx.x;
    if (i < n) dst[i] = (__bf16)src[i];
}

// ---------------------------------------------------------------------------
// GEMM: C[m,n] = sum_k A[m,k] * W[n,k]   (A: MxK bf16, W: NxK bf16, C: MxN f32)
// Wave tile 32x32 (2x2 wmma), workgroup 8 waves -> 64(M) x 128(N) tile.
// ---------------------------------------------------------------------------
__global__ __launch_bounds__(256)
void gemm_bf16_nt_kernel(const __bf16* __restrict__ A, const __bf16* __restrict__ W,
                         float* __restrict__ C, int M, int N, int K) {
    const int lane = threadIdx.x & 31;
    const int wave = threadIdx.x >> 5;
    const int m0 = blockIdx.y * 64 + (wave >> 2) * 32;
    const int n0 = blockIdx.x * 128 + (wave & 3) * 32;
    const int r16 = lane & 15;
    const int hi = lane >> 4;
    const int a_koff = hi * 16;
    const int b_koff = hi * 16;

    v8f acc00 = vzero8(), acc01 = vzero8(), acc10 = vzero8(), acc11 = vzero8();

    const __bf16* aRow0 = A + (size_t)(m0 + r16) * K;
    const __bf16* aRow1 = A + (size_t)(m0 + 16 + r16) * K;
    const __bf16* wRow0 = W + (size_t)(n0 + r16) * K;
    const __bf16* wRow1 = W + (size_t)(n0 + 16 + r16) * K;

    for (int k0 = 0; k0 < K; k0 += 32) {
        __builtin_prefetch(wRow0 + k0 + 256, 0, 1);   // global_prefetch_b8
        v16bf a0 = make_a_frag(aRow0 + k0 + a_koff);
        v16bf a1 = make_a_frag(aRow1 + k0 + a_koff);
        v16bf b0 = make_frag16(wRow0 + k0 + b_koff);
        v16bf b1 = make_frag16(wRow1 + k0 + b_koff);
        acc00 = wmma_bf16(a0, b0, acc00);
        acc01 = wmma_bf16(a0, b1, acc01);
        acc10 = wmma_bf16(a1, b0, acc10);
        acc11 = wmma_bf16(a1, b1, acc11);
    }

    // C/D layout: lanes 0-15: M=vgpr, N=lane; lanes 16-31: M=8+vgpr, N=lane-16
#pragma unroll
    for (int v = 0; v < 8; ++v) {
        int rA = m0 + v + 8 * hi;
        C[(size_t)rA * N + n0 + r16]             = acc00[v];
        C[(size_t)rA * N + n0 + 16 + r16]        = acc01[v];
        C[(size_t)(rA + 16) * N + n0 + r16]      = acc10[v];
        C[(size_t)(rA + 16) * N + n0 + 16 + r16] = acc11[v];
    }
}

// ---------------------------------------------------------------------------
// RMS-norm (over DIM) + partial RoPE (first 64 of each head) + scatter to
// head-major bf16 [HEADS][rows][HD]. One block per token row. `scale` folds
// the softmax 1/sqrt(HD) into Q (rope is linear, so scaling commutes).
// ---------------------------------------------------------------------------
__global__ __launch_bounds__(256)
void norm_rope_kernel(const float* __restrict__ X, const float* __restrict__ w,
                      const float* __restrict__ rope, __bf16* __restrict__ out,
                      int rows, float scale) {
    const int s = blockIdx.x;
    const float* x = X + (size_t)s * DIMC;
    __shared__ float red[8];

    float ss = 0.0f;
    for (int i = threadIdx.x; i < DIMC; i += 256) { float v = x[i]; ss += v * v; }
#pragma unroll
    for (int off = 16; off >= 1; off >>= 1) ss += __shfl_xor(ss, off, 32);
    if ((threadIdx.x & 31) == 0) red[threadIdx.x >> 5] = ss;
    __syncthreads();
    float tot = 0.0f;
#pragma unroll
    for (int i = 0; i < 8; ++i) tot += red[i];
    const float rn = rsqrtf(tot * (1.0f / DIMC) + EPSF) * scale;

    const float* rp = rope + (size_t)s * 128;   // [32][2][2] per token
    for (int p = threadIdx.x; p < DIMC / 2; p += 256) {
        int h = p >> 6;          // head
        int t = p & 63;          // pair index within head (0..63)
        int i0 = h * HDC + 2 * t;
        float x0 = x[i0]     * rn * w[i0];
        float x1 = x[i0 + 1] * rn * w[i0 + 1];
        float o0, o1;
        if (t < 32) {
            float r00 = rp[t * 4 + 0], r01 = rp[t * 4 + 1];
            float r10 = rp[t * 4 + 2], r11 = rp[t * 4 + 3];
            o0 = r00 * x0 + r01 * x1;
            o1 = r10 * x0 + r11 * x1;
        } else {
            o0 = x0; o1 = x1;
        }
        __bf16* dst = out + ((size_t)h * rows + s) * HDC + 2 * t;
        dst[0] = (__bf16)o0;
        dst[1] = (__bf16)o1;
    }
}

// V: f32 [S][DIM] -> bf16 head-major [HEADS][S][HD]
__global__ __launch_bounds__(256)
void v_scatter_kernel(const float* __restrict__ V32, __bf16* __restrict__ Vh, int rows) {
    int i = blockIdx.x * 256 + threadIdx.x;
    int n = rows * DIMC;
    if (i >= n) return;
    int sr = i / DIMC, c = i - sr * DIMC;
    int h = c >> 7, d = c & 127;
    Vh[((size_t)h * rows + sr) * HDC + d] = (__bf16)V32[i];
}

// ---------------------------------------------------------------------------
// Flash attention (no mask: query rows < BLOCKQ attend all S_TOK keys).
// Block = (head, 128-query chunk); 8 waves x 16 queries each.
// Per 32-key tile: 8 wmma (scores, K from global) + 8 wmma (P*V, V via LDS
// transpose). Online softmax with half-wave shuffle row-reductions.
// Output: bf16 [BLOCKQ][DIM] (token-major, head dims interleaved) for Wo GEMM.
// ---------------------------------------------------------------------------
__global__ __launch_bounds__(256)
void attn_kernel(const __bf16* __restrict__ Qh, const __bf16* __restrict__ Kh,
                 const __bf16* __restrict__ Vh, __bf16* __restrict__ Ob) {
    __shared__ __bf16 Vt[HDC * 40];      // V^T tile: [d][key], key-stride 40
    __shared__ __bf16 Pw[8 * 16 * 32];   // per-wave P scratch (16 rows x 32 keys)

    const int lane = threadIdx.x & 31;
    const int wave = threadIdx.x >> 5;
    const int head = blockIdx.y;
    const int q0   = blockIdx.x * 128 + wave * 16;

    const __bf16* Qb = Qh + ((size_t)head * BLOCKQ + q0) * HDC;
    const __bf16* Kb = Kh + (size_t)head * S_TOK * HDC;
    const __bf16* Vb = Vh + (size_t)head * S_TOK * HDC;

    const int r16    = lane & 15;
    const int hi     = lane >> 4;
    const int a_koff = hi * 16;
    const int b_koff = hi * 16;

    // Q fragments for HD=128 (4 chunks of K=32), loaded once.
    v16bf qa[4];
#pragma unroll
    for (int c = 0; c < 4; ++c)
        qa[c] = make_a_frag(Qb + (size_t)r16 * HDC + c * 32 + a_koff);

    v8f o[8];
    float mrun[8], lrun[8];
#pragma unroll
    for (int j = 0; j < 8; ++j) o[j] = vzero8();
#pragma unroll
    for (int v = 0; v < 8; ++v) { mrun[v] = -1e30f; lrun[v] = 0.0f; }

    __bf16* Pws = Pw + wave * 512;

    for (int kt = 0; kt < S_TOK; kt += 32) {
        __syncthreads();
        // Cooperative V tile transpose: Vt[d][k] = V[kt+k][d]
        {
            int k  = threadIdx.x >> 3;          // 0..31 key within tile
            int dc = (threadIdx.x & 7) * 16;    // d-chunk
            const __bf16* sp = Vb + (size_t)(kt + k) * HDC + dc;
            v8bf lo = *(const v8bf*)sp;
            v8bf hh = *(const v8bf*)(sp + 8);
#pragma unroll
            for (int i = 0; i < 8; ++i) {
                Vt[(dc + i) * 40 + k]     = lo[i];
                Vt[(dc + 8 + i) * 40 + k] = hh[i];
            }
        }
        __syncthreads();

        // ---- scores: S = Q * K^T (Q pre-scaled) ----
        v8f s0 = vzero8(), s1 = vzero8();
#pragma unroll
        for (int c = 0; c < 4; ++c) {
            v16bf b0 = make_frag16(Kb + (size_t)(kt + r16) * HDC + c * 32 + b_koff);
            v16bf b1 = make_frag16(Kb + (size_t)(kt + 16 + r16) * HDC + c * 32 + b_koff);
            s0 = wmma_bf16(qa[c], b0, s0);
            s1 = wmma_bf16(qa[c], b1, s1);
        }

        // ---- online softmax (row = vgpr index + 8*hi, cols across 16 lanes) ----
        float p0a[8], p1a[8];
#pragma unroll
        for (int v = 0; v < 8; ++v) {
            float mx = fmaxf(s0[v], s1[v]);
            mx = fmaxf(mx, __shfl_xor(mx, 1, 32));
            mx = fmaxf(mx, __shfl_xor(mx, 2, 32));
            mx = fmaxf(mx, __shfl_xor(mx, 4, 32));
            mx = fmaxf(mx, __shfl_xor(mx, 8, 32));
            float nm = fmaxf(mrun[v], mx);
            float al = __expf(mrun[v] - nm);
            float e0 = __expf(s0[v] - nm);
            float e1 = __expf(s1[v] - nm);
            float rs = e0 + e1;
            rs += __shfl_xor(rs, 1, 32);
            rs += __shfl_xor(rs, 2, 32);
            rs += __shfl_xor(rs, 4, 32);
            rs += __shfl_xor(rs, 8, 32);
            lrun[v] = lrun[v] * al + rs;
            mrun[v] = nm;
            p0a[v] = e0; p1a[v] = e1;
#pragma unroll
            for (int j = 0; j < 8; ++j) o[j][v] *= al;
        }

        // ---- P: C-layout -> A-layout via wave-private LDS ----
#pragma unroll
        for (int v = 0; v < 8; ++v) {
            int row = v + 8 * hi;
            Pws[row * 32 + r16]      = (__bf16)p0a[v];
            Pws[row * 32 + 16 + r16] = (__bf16)p1a[v];
        }
        v16bf pa = make_a_frag(Pws + r16 * 32 + a_koff);

        // ---- O += P * V (B frags contiguous from transposed LDS tile) ----
#pragma unroll
        for (int j = 0; j < 8; ++j) {
            v16bf bv = make_frag16(Vt + (j * 16 + r16) * 40 + b_koff);
            o[j] = wmma_bf16(pa, bv, o[j]);
        }
    }

    // normalize & store (token-major [q][DIM], head slice interleaved)
#pragma unroll
    for (int v = 0; v < 8; ++v) {
        float inv = 1.0f / lrun[v];
        int row = q0 + v + 8 * hi;
#pragma unroll
        for (int j = 0; j < 8; ++j)
            Ob[(size_t)row * DIMC + head * HDC + j * 16 + r16] = (__bf16)(o[j][v] * inv);
    }
}

// ---------------------------------------------------------------------------
// Host-side launcher
// ---------------------------------------------------------------------------
extern "C" void kernel_launch(void* const* d_in, const int* in_sizes, int n_in,
                              void* d_out, int out_size, void* d_ws, size_t ws_size,
                              hipStream_t stream) {
    (void)in_sizes; (void)n_in; (void)out_size; (void)ws_size;

    const float* X    = (const float*)d_in[0];  // [1,S,DIM]
    const float* rope = (const float*)d_in[1];  // [1,S,1,32,2,2]
    const float* Wq   = (const float*)d_in[2];
    const float* Wk   = (const float*)d_in[3];
    const float* Wv   = (const float*)d_in[4];
    const float* Wo   = (const float*)d_in[5];
    const float* qw   = (const float*)d_in[6];
    const float* kw   = (const float*)d_in[7];
    // d_in[8] proj_down, d_in[9] proj_up: dead code — LoRA only touches rows
    // >= BLOCKQ, which are sliced out of the output.

    char* base = (char*)d_ws;
    size_t off = 0;
    auto carve = [&](size_t bytes) -> char* {
        char* p = base + off;
        off += (bytes + 255) & ~(size_t)255;
        return p;
    };

    __bf16* Xb  = (__bf16*)carve((size_t)S_TOK * DIMC * 2);
    __bf16* Wqb = (__bf16*)carve((size_t)DIMC * DIMC * 2);
    __bf16* Wkb = (__bf16*)carve((size_t)DIMC * DIMC * 2);
    __bf16* Wvb = (__bf16*)carve((size_t)DIMC * DIMC * 2);
    __bf16* Wob = (__bf16*)carve((size_t)DIMC * DIMC * 2);
    float*  Q32 = (float*)carve((size_t)BLOCKQ * DIMC * 4);
    float*  K32 = (float*)carve((size_t)S_TOK * DIMC * 4);
    float*  V32 = (float*)carve((size_t)S_TOK * DIMC * 4);
    __bf16* Qh  = (__bf16*)carve((size_t)NHEADS * BLOCKQ * HDC * 2);
    __bf16* Kh  = (__bf16*)carve((size_t)NHEADS * S_TOK * HDC * 2);
    __bf16* Vh  = (__bf16*)carve((size_t)NHEADS * S_TOK * HDC * 2);
    __bf16* Ob  = (__bf16*)carve((size_t)BLOCKQ * DIMC * 2);

    // 1) convert inputs to bf16
    int nX = S_TOK * DIMC;
    int nW = DIMC * DIMC;
    f32_to_bf16_kernel<<<(nX + 255) / 256, 256, 0, stream>>>(X, Xb, nX);
    f32_to_bf16_kernel<<<(nW + 255) / 256, 256, 0, stream>>>(Wq, Wqb, nW);
    f32_to_bf16_kernel<<<(nW + 255) / 256, 256, 0, stream>>>(Wk, Wkb, nW);
    f32_to_bf16_kernel<<<(nW + 255) / 256, 256, 0, stream>>>(Wv, Wvb, nW);
    f32_to_bf16_kernel<<<(nW + 255) / 256, 256, 0, stream>>>(Wo, Wob, nW);

    // 2) QKV projections (Q only needs the first BLOCKQ rows)
    dim3 gq(DIMC / 128, BLOCKQ / 64);   // (20, 32)
    dim3 gk(DIMC / 128, S_TOK / 64);    // (20, 48)
    gemm_bf16_nt_kernel<<<gq, 256, 0, stream>>>(Xb, Wqb, Q32, BLOCKQ, DIMC, DIMC);
    gemm_bf16_nt_kernel<<<gk, 256, 0, stream>>>(Xb, Wkb, K32, S_TOK, DIMC, DIMC);
    gemm_bf16_nt_kernel<<<gk, 256, 0, stream>>>(Xb, Wvb, V32, S_TOK, DIMC, DIMC);

    // 3) RMS-norm + partial RoPE -> head-major bf16 (softmax scale folded in Q)
    norm_rope_kernel<<<BLOCKQ, 256, 0, stream>>>(Q32, qw, rope, Qh, BLOCKQ, SCALEF);
    norm_rope_kernel<<<S_TOK, 256, 0, stream>>>(K32, kw, rope, Kh, S_TOK, 1.0f);
    v_scatter_kernel<<<(nX + 255) / 256, 256, 0, stream>>>(V32, Vh, S_TOK);

    // 4) flash attention
    dim3 ga(BLOCKQ / 128, NHEADS);      // (16, 20)
    attn_kernel<<<ga, 256, 0, stream>>>(Qh, Kh, Vh, Ob);

    // 5) output projection straight into d_out (f32)
    dim3 go(DIMC / 128, BLOCKQ / 64);
    gemm_bf16_nt_kernel<<<go, 256, 0, stream>>>(Ob, Wob, (float*)d_out, BLOCKQ, DIMC, DIMC);
}
